// Cross_Covariance_Attention_49701361549930
// MI455X (gfx1250) — matrix operand order repaired and checked by reading, not compiled, not run on hardware
//
#include <hip/hip_runtime.h>
#include <hip/hip_bf16.h>

// Problem constants
#define B_    16
#define N_    3136
#define C_    768
#define H_    8
#define D_    96
#define M_    (B_*N_)     // 50176 tokens total
#define QKV_  (3*C_)      // 2304
#define BH_   (B_*H_)     // 128

typedef __attribute__((ext_vector_type(16))) __bf16 v16bf;
typedef __attribute__((ext_vector_type(8)))  float  v8f;

union FragU { uint4 u[2]; v16bf v; };
union PackU { unsigned short us[8]; uint4 u4; };

__device__ __forceinline__ unsigned short f2bf(float x) {
  unsigned int u = __float_as_uint(x);
  u += 0x7fffu + ((u >> 16) & 1u);          // round-to-nearest-even
  return (unsigned short)(u >> 16);
}
__device__ __forceinline__ float bf2f(unsigned short h) {
  return __uint_as_float(((unsigned int)h) << 16);
}

__device__ __forceinline__ v8f wmma_bf16(v16bf a, v16bf b, v8f c) {
  return __builtin_amdgcn_wmma_f32_16x16x32_bf16(false, a, false, b, (short)0, c,
                                                 false, false);
}

// A fragment: A row-major [16 rows][K], base already points at row m0.
// Lane L: row = L&15; K chunks {base..base+7, base+16..base+23}, base = (L<16?0:8).
__device__ __forceinline__ v16bf load_a_frag(const unsigned short* __restrict__ A,
                                             int lda, int k0, int lane) {
  const unsigned short* p = A + (long long)(lane & 15) * lda + k0 + ((lane < 16) ? 0 : 8);
  FragU f;
  f.u[0] = *(const uint4*)(p);
  f.u[1] = *(const uint4*)(p + 16);
  return f.v;
}
// B fragment from transposed storage Bt[Ncol][K], base already points at col n0.
// Lane L: col = L&15; K chunks {base..base+7, base+16..base+23}, base = (L<16?0:8).
__device__ __forceinline__ v16bf load_b_frag(const unsigned short* __restrict__ Bt,
                                             int ldb, int k0, int lane) {
  const unsigned short* p = Bt + (long long)(lane & 15) * ldb + k0 + ((lane < 16) ? 0 : 8);
  FragU f;
  f.u[0] = *(const uint4*)(p);
  f.u[1] = *(const uint4*)(p + 16);
  return f.v;
}

// ---------------- conversion kernels ----------------
__global__ void cvt_bf16(const float* __restrict__ src, unsigned short* __restrict__ dst,
                         long long n) {
  long long i = (long long)blockIdx.x * blockDim.x + threadIdx.x;
  if (i < n) dst[i] = f2bf(src[i]);
}
// W [rows][cols] row-major  ->  Wt [cols][rows]  (bf16)
__global__ void cvt_bf16_T(const float* __restrict__ W, unsigned short* __restrict__ Wt,
                           int rows, int cols) {
  long long i = (long long)blockIdx.x * blockDim.x + threadIdx.x;
  if (i < (long long)rows * cols) {
    int r = (int)(i / cols), c = (int)(i % cols);
    Wt[(long long)c * rows + r] = f2bf(W[i]);
  }
}

// ---------------- stage 1: qkv = x @ W_qkv, scatter to q/k (channel-major) and v (token-major)
__global__ __launch_bounds__(128) void qkv_gemm(
    const unsigned short* __restrict__ xb,   // [M_][C_] bf16
    const unsigned short* __restrict__ wT,   // [QKV_][C_] bf16 (transposed weights)
    unsigned short* __restrict__ qb,         // [BH_][D_][N_]
    unsigned short* __restrict__ kb,         // [BH_][D_][N_]
    unsigned short* __restrict__ vb)         // [BH_][N_][D_]
{
  __shared__ float lds[4][16][17];
  const int lane = threadIdx.x & 31;
  const int wid  = threadIdx.x >> 5;
  const int w    = blockIdx.x * 4 + wid;         // 3136*36 waves
  const int m0   = (w / 36) * 16;
  const int c0   = (w % 36) * 64;

  v8f acc[4] = { {}, {}, {}, {} };
  const unsigned short* ap = xb + (long long)m0 * C_;
  for (int k0 = 0; k0 < C_; k0 += 32) {
    v16bf a = load_a_frag(ap, C_, k0, lane);
#pragma unroll
    for (int t = 0; t < 4; ++t) {
      v16bf b = load_b_frag(wT + (long long)(c0 + t * 16) * C_, C_, k0, lane);
      acc[t] = wmma_bf16(a, b, acc[t]);
    }
  }

  const int bidx = m0 / N_;
  const int n0   = m0 % N_;            // 16 | 3136 -> no batch straddle
  const int col  = lane & 15;
  const int rsel = (lane < 16) ? 0 : 8;
#pragma unroll
  for (int t = 0; t < 4; ++t) {
    const int cg   = c0 + t * 16;
    const int sec  = cg / C_;          // 0=q,1=k,2=v (strip never crosses a section)
    const int cc   = cg % C_;
    const int head = cc / D_;
    const int ch0  = cc % D_;          // multiple of 16
    const long long bh = (long long)bidx * H_ + head;
    if (sec < 2) {
      // channel-major: lane already holds 8 consecutive tokens of one channel
      PackU pk;
#pragma unroll
      for (int j = 0; j < 8; ++j) pk.us[j] = f2bf(acc[t][j]);
      unsigned short* dst = (sec == 0 ? qb : kb) +
                            (bh * D_ + ch0 + col) * (long long)N_ + n0 + rsel;
      *(uint4*)dst = pk.u4;
    } else {
      // token-major v: transpose 16x16 tile through wave-private LDS
#pragma unroll
      for (int j = 0; j < 8; ++j) lds[wid][rsel + j][col] = acc[t][j];
      asm volatile("s_wait_dscnt 0" ::: "memory");
      PackU pk;
#pragma unroll
      for (int i = 0; i < 8; ++i) pk.us[i] = f2bf(lds[wid][col][rsel + i]);
      unsigned short* dst = vb + (bh * N_ + n0 + col) * (long long)D_ + ch0 + rsel;
      *(uint4*)dst = pk.u4;
      asm volatile("s_wait_dscnt 0" ::: "memory");
    }
  }
}

// ---------------- stage 2a: reciprocal L2 norms of q/k channels (over N tokens)
__global__ __launch_bounds__(128) void norms_k(
    const unsigned short* __restrict__ qb, const unsigned short* __restrict__ kb,
    float* __restrict__ rq, float* __restrict__ rk)
{
  const int lane = threadIdx.x & 31;
  const int row  = blockIdx.x * 4 + (threadIdx.x >> 5);   // 0 .. 2*BH_*D_-1
  const bool isQ = row < BH_ * D_;
  const int  r   = isQ ? row : row - BH_ * D_;
  const unsigned short* p = (isQ ? qb : kb) + (long long)r * N_;
  float s = 0.f;
  for (int i = lane; i < N_; i += 32) { float v = bf2f(p[i]); s += v * v; }
#pragma unroll
  for (int off = 16; off > 0; off >>= 1) s += __shfl_xor(s, off, 32);
  if (lane == 0) {
    float rr = 1.0f / fmaxf(sqrtf(s), 1e-12f);
    (isQ ? rq : rk)[r] = rr;
  }
}

// ---------------- stage 2b: attn_raw[bh][c][e] = sum_n k[c,n]*q[e,n]
__global__ __launch_bounds__(128) void attn_gemm(
    const unsigned short* __restrict__ kb, const unsigned short* __restrict__ qb,
    float* __restrict__ attn_raw)
{
  const int lane = threadIdx.x & 31;
  const int w    = blockIdx.x * 4 + (threadIdx.x >> 5);   // 128*36 waves
  const int bh   = w / 36;
  const int t    = w % 36;
  const int c0   = (t / 6) * 16, e0 = (t % 6) * 16;
  const unsigned short* kp = kb + (long long)bh * D_ * N_ + (long long)c0 * N_;
  const unsigned short* qp = qb + (long long)bh * D_ * N_ + (long long)e0 * N_;
  v8f acc = {};
  for (int k0 = 0; k0 < N_; k0 += 32)
    acc = wmma_bf16(load_a_frag(kp, N_, k0, lane), load_b_frag(qp, N_, k0, lane), acc);
  float* dst = attn_raw + (long long)bh * (D_ * D_);
  const int col = lane & 15, rsel = (lane < 16) ? 0 : 8;
#pragma unroll
  for (int j = 0; j < 8; ++j) dst[(c0 + rsel + j) * D_ + e0 + col] = acc[j];
}

// ---------------- stage 2c: fold norms + temp, row softmax, emit bf16 attn
__global__ void softmax_k(float* __restrict__ attn_raw,
                          const float* __restrict__ rq, const float* __restrict__ rk,
                          const float* __restrict__ temp,
                          unsigned short* __restrict__ attnb)
{
  const int bh = blockIdx.x;
  const int c  = threadIdx.x;        // 96 threads
  const float tv  = temp[bh & 7];
  const float rkc = rk[bh * D_ + c];
  float* row = attn_raw + (long long)bh * (D_ * D_) + c * D_;
  const float* rqb = rq + bh * D_;
  float mx = -3.4e38f;
  for (int e = 0; e < D_; ++e) mx = fmaxf(mx, row[e] * rkc * rqb[e] * tv);
  float s = 0.f;
  for (int e = 0; e < D_; ++e) {
    float x = __expf(row[e] * rkc * rqb[e] * tv - mx);
    s += x; row[e] = x;
  }
  const float inv = 1.0f / s;
  unsigned short* orow = attnb + (long long)bh * (D_ * D_) + c * D_;
  for (int e = 0; e < D_; ++e) orow[e] = f2bf(row[e] * inv);
}

// ---------------- stage 3: out[c][n] = attn @ v, written directly into y[B,N,C] (bf16)
__global__ __launch_bounds__(128) void out_gemm(
    const unsigned short* __restrict__ attnb,  // [BH_][D_][D_]
    const unsigned short* __restrict__ vb,     // [BH_][N_][D_]
    unsigned short* __restrict__ yb)           // [M_][C_]
{
  const int lane = threadIdx.x & 31;
  const int w    = blockIdx.x * 4 + (threadIdx.x >> 5);   // 128*6*49 waves
  const int bh   = w / 294;
  const int r    = w % 294;
  const int c0   = (r / 49) * 16;
  const int n0   = (r % 49) * 64;
  const unsigned short* ap = attnb + (long long)bh * (D_ * D_) + (long long)c0 * D_;
  const unsigned short* bp = vb + (long long)bh * N_ * D_;
  v8f acc[4] = { {}, {}, {}, {} };
  for (int k0 = 0; k0 < D_; k0 += 32) {
    v16bf a = load_a_frag(ap, D_, k0, lane);
#pragma unroll
    for (int t = 0; t < 4; ++t)
      acc[t] = wmma_bf16(a, load_b_frag(bp + (long long)(n0 + t * 16) * D_, D_, k0, lane),
                         acc[t]);
  }
  const int b_ = bh >> 3, h_ = bh & 7;
  const int col = lane & 15, rsel = (lane < 16) ? 0 : 8;
#pragma unroll
  for (int t = 0; t < 4; ++t) {
    PackU pk;
#pragma unroll
    for (int j = 0; j < 8; ++j) pk.us[j] = f2bf(acc[t][j]);
    long long idx = ((long long)b_ * N_ + n0 + t * 16 + col) * C_ + h_ * D_ + c0 + rsel;
    *(uint4*)(yb + idx) = pk.u4;
  }
}

// ---------------- stage 4: final = y @ W_proj + b_proj (fp32 out, LDS transpose for rows)
__global__ __launch_bounds__(128) void proj_gemm(
    const unsigned short* __restrict__ yb,   // [M_][C_]
    const unsigned short* __restrict__ wT,   // [C_][C_] transposed
    const float* __restrict__ bias,
    float* __restrict__ out)                 // [M_][C_]
{
  __shared__ float lds[4][16][17];
  const int lane = threadIdx.x & 31;
  const int wid  = threadIdx.x >> 5;
  const int w    = blockIdx.x * 4 + wid;     // 3136*12 waves
  const int m0   = (w / 12) * 16;
  const int c0   = (w % 12) * 64;
  v8f acc[4] = { {}, {}, {}, {} };
  const unsigned short* ap = yb + (long long)m0 * C_;
  for (int k0 = 0; k0 < C_; k0 += 32) {
    v16bf a = load_a_frag(ap, C_, k0, lane);
#pragma unroll
    for (int t = 0; t < 4; ++t)
      acc[t] = wmma_bf16(a, load_b_frag(wT + (long long)(c0 + t * 16) * C_, C_, k0, lane),
                         acc[t]);
  }
  const int col = lane & 15, rsel = (lane < 16) ? 0 : 8;
#pragma unroll
  for (int t = 0; t < 4; ++t) {
    const float bv = bias[c0 + t * 16 + col];
#pragma unroll
    for (int j = 0; j < 8; ++j) lds[wid][rsel + j][col] = acc[t][j] + bv;
    asm volatile("s_wait_dscnt 0" ::: "memory");
    float4 f0, f1;
    f0.x = lds[wid][col][rsel + 0]; f0.y = lds[wid][col][rsel + 1];
    f0.z = lds[wid][col][rsel + 2]; f0.w = lds[wid][col][rsel + 3];
    f1.x = lds[wid][col][rsel + 4]; f1.y = lds[wid][col][rsel + 5];
    f1.z = lds[wid][col][rsel + 6]; f1.w = lds[wid][col][rsel + 7];
    float* dst = out + (long long)(m0 + col) * C_ + c0 + t * 16 + rsel;
    *(float4*)dst       = f0;
    *(float4*)(dst + 4) = f1;
    asm volatile("s_wait_dscnt 0" ::: "memory");
  }
}

extern "C" void kernel_launch(void* const* d_in, const int* in_sizes, int n_in,
                              void* d_out, int out_size, void* d_ws, size_t ws_size,
                              hipStream_t stream) {
  (void)in_sizes; (void)n_in; (void)out_size; (void)ws_size;
  const float* x      = (const float*)d_in[0];
  const float* W_qkv  = (const float*)d_in[1];
  const float* temp   = (const float*)d_in[2];
  const float* W_proj = (const float*)d_in[3];
  const float* b_proj = (const float*)d_in[4];
  float* out = (float*)d_out;

  char* p = (char*)d_ws;
  auto alloc = [&](size_t bytes) -> char* {
    char* r = p; p += (bytes + 255) & ~(size_t)255; return r;
  };
  unsigned short* xb    = (unsigned short*)alloc((size_t)M_ * C_ * 2);
  unsigned short* wqT   = (unsigned short*)alloc((size_t)QKV_ * C_ * 2);
  unsigned short* wpT   = (unsigned short*)alloc((size_t)C_ * C_ * 2);
  unsigned short* qb    = (unsigned short*)alloc((size_t)BH_ * D_ * N_ * 2);
  unsigned short* kb    = (unsigned short*)alloc((size_t)BH_ * D_ * N_ * 2);
  unsigned short* vb    = (unsigned short*)alloc((size_t)BH_ * N_ * D_ * 2);
  unsigned short* yb    = (unsigned short*)alloc((size_t)M_ * C_ * 2);
  float*          araw  = (float*)alloc((size_t)BH_ * D_ * D_ * 4);
  unsigned short* attnb = (unsigned short*)alloc((size_t)BH_ * D_ * D_ * 2);
  float*          rqv   = (float*)alloc((size_t)BH_ * D_ * 4);
  float*          rkv   = (float*)alloc((size_t)BH_ * D_ * 4);

  const long long nx = (long long)M_ * C_;
  cvt_bf16<<<(int)((nx + 255) / 256), 256, 0, stream>>>(x, xb, nx);
  cvt_bf16_T<<<(C_ * QKV_ + 255) / 256, 256, 0, stream>>>(W_qkv, wqT, C_, QKV_);
  cvt_bf16_T<<<(C_ * C_ + 255) / 256, 256, 0, stream>>>(W_proj, wpT, C_, C_);

  qkv_gemm<<<(M_ / 16) * 36 / 4, 128, 0, stream>>>(xb, wqT, qb, kb, vb);     // 28224 blocks
  norms_k <<<2 * BH_ * D_ / 4, 128, 0, stream>>>(qb, kb, rqv, rkv);          // 6144 blocks
  attn_gemm<<<BH_ * 36 / 4, 128, 0, stream>>>(kb, qb, araw);                 // 1152 blocks
  softmax_k<<<BH_, D_, 0, stream>>>(araw, rqv, rkv, temp, attnb);
  out_gemm<<<BH_ * 6 * 49 / 4, 128, 0, stream>>>(attnb, vb, yb);             // 9408 blocks
  proj_gemm<<<(M_ / 16) * 12 / 4, 128, 0, stream>>>(yb, wpT, b_proj, out);   // 9408 blocks
}
